// FullMHA_5471788335671
// MI455X (gfx1250) — compile-verified, matrix operand order checked by reading, not run on hardware
//
#include <hip/hip_runtime.h>
#include <hip/hip_bf16.h>

// ---------------- problem constants ----------------
#define DIMC    2048
#define NHEADS  16
#define HEADD   128
#define SEQ     512
#define NBATCH  16
#define QK_SCALE 0.08838834764831845f   // 1/sqrt(128)

typedef __bf16 bf16_t;
typedef bf16_t v16bf __attribute__((ext_vector_type(16)));
typedef bf16_t v8bf  __attribute__((ext_vector_type(8)));
typedef float  v8f   __attribute__((ext_vector_type(8)));

union Frag16 { v16bf v; v8bf h[2]; };

// Load a 16x32 bf16 WMMA A-fragment (or B-fragment from a column-major /
// transposed tile). Per CDNA5 ISA: lane l -> row m = l&15,
// khalf = l>>4; vector elements 0..7 hold k = khalf*8 + 0..7, elements
// 8..15 hold k = 16 + khalf*8 + 0..7  => two contiguous 16-byte runs.
template <typename P>
__device__ __forceinline__ v16bf load_frag16(const P* base, int ld) {
  const int l  = threadIdx.x & 31;
  const int m  = l & 15;
  const int kh = l >> 4;
  const bf16_t* p = (const bf16_t*)base + (size_t)m * ld + kh * 8;
  Frag16 f;
  f.h[0] = *(const v8bf*)(p);
  f.h[1] = *(const v8bf*)(p + 16);
  return f.v;
}

__device__ __forceinline__ v8f wmma_bf16(v16bf a, v16bf b, v8f c) {
  return __builtin_amdgcn_wmma_f32_16x16x32_bf16(
      /*neg_a=*/false, a, /*neg_b=*/false, b,
      /*c_mod=*/(short)0, c, /*reuse_a=*/false, /*reuse_b=*/false);
}

// =====================================================================
// Kernel 1: QKV projection (X[8192,2048] @ Wqkv[2048,6144]) with fused
// RoPE + scale + bf16 scatter into q/k/v [B,H,T,hd] buffers.
// Block = 256 threads = 8 waves; block tile 128x128, K-step 32.
// Wave (wy in 0..3, wx in 0..1) owns a 32x64 patch = 2x4 WMMA tiles.
// =====================================================================
__global__ __launch_bounds__(256) void qkv_rope_kernel(
    const float* __restrict__ X, const float* __restrict__ W,
    bf16_t* __restrict__ qb, bf16_t* __restrict__ kb, bf16_t* __restrict__ vb)
{
  const int K = DIMC, N = 3 * DIMC;
  __shared__ __align__(16) bf16_t As[128][40];   // row-major   [m][k]
  __shared__ __align__(16) bf16_t Bs[128][40];   // transposed  [n][k]

  const int tid  = threadIdx.x;
  const int wid  = tid >> 5;
  const int lane = tid & 31;
  const int wy = wid & 3, wx = wid >> 2;
  const int row0 = blockIdx.y * 128;
  const int col0 = blockIdx.x * 128;

  v8f acc[2][4];
  for (int i = 0; i < 2; ++i)
    for (int j = 0; j < 4; ++j)
      acc[i][j] = (v8f)(0.f);

  for (int k0 = 0; k0 < K; k0 += 32) {
    // ---- stage A: 128x32 fp32 -> bf16 (4 float4 per thread) ----
    for (int i = 0; i < 4; ++i) {
      int idx = tid + i * 256;        // 0..1023
      int r   = idx >> 3;             // 8 float4 per 32-wide row
      int c4  = idx & 7;
      float4 f = *(const float4*)(X + (size_t)(row0 + r) * K + k0 + c4 * 4);
      bf16_t* dst = &As[r][c4 * 4];
      dst[0] = (bf16_t)f.x; dst[1] = (bf16_t)f.y;
      dst[2] = (bf16_t)f.z; dst[3] = (bf16_t)f.w;
    }
    // ---- stage B: 32x128 fp32 -> bf16 transposed ----
    for (int i = 0; i < 4; ++i) {
      int idx = tid + i * 256;
      int kr  = idx >> 5;             // 32 float4 per 128-wide row
      int c4  = idx & 31;
      float4 f = *(const float4*)(W + (size_t)(k0 + kr) * N + col0 + c4 * 4);
      Bs[c4 * 4 + 0][kr] = (bf16_t)f.x;
      Bs[c4 * 4 + 1][kr] = (bf16_t)f.y;
      Bs[c4 * 4 + 2][kr] = (bf16_t)f.z;
      Bs[c4 * 4 + 3][kr] = (bf16_t)f.w;
    }
    __syncthreads();

    v16bf a0 = load_frag16(&As[wy * 32][0],      40);
    v16bf a1 = load_frag16(&As[wy * 32 + 16][0], 40);
    for (int j = 0; j < 4; ++j) {
      v16bf b = load_frag16(&Bs[wx * 64 + j * 16][0], 40);
      acc[0][j] = wmma_bf16(a0, b, acc[0][j]);
      acc[1][j] = wmma_bf16(a1, b, acc[1][j]);
    }
    __syncthreads();
  }

  // ---- epilogue: RoPE + scale + bf16 scatter ----
  const int nlo  = lane & 15;
  const int half = lane >> 4;
  const int s = col0 / DIMC;                 // 0=q, 1=k, 2=v (block-uniform)
  const int h = (col0 % DIMC) / HEADD;       // head (block-uniform)
  for (int i = 0; i < 2; ++i) {
    for (int j = 0; j < 4; ++j) {
      const int d = wx * 64 + j * 16 + nlo;            // dim within head
      const bool do_rope = (s < 2) && (wx == 0) && (j == 0);  // d<16, uniform
      for (int r = 0; r < 8; ++r) {
        int mg = row0 + wy * 32 + i * 16 + r + half * 8;
        int bidx = mg >> 9;
        int t    = mg & (SEQ - 1);
        float val = acc[i][j][r];
        if (do_rope) {
          float other = __shfl_xor(val, 8, 32);        // pair d <-> d+8
          int fi = d & 7;
          float ang = (float)t * __powf(10000.f, -(float)fi * 0.125f);
          float sn, cs;
          __sincosf(ang, &sn, &cs);
          val = (d < 8) ? (val * cs - other * sn) : (val * cs + other * sn);
        }
        if (s == 0) val *= QK_SCALE;
        bf16_t o = (bf16_t)val;
        size_t idx = ((size_t)((bidx * NHEADS + h) * SEQ + t)) * HEADD + d;
        if (s == 0)      qb[idx] = o;
        else if (s == 1) kb[idx] = o;
        else             vb[idx] = o;
      }
    }
  }
}

// =====================================================================
// Kernel 2: causal flash attention. Block = (b, h, 128-query tile),
// 8 waves x 16 query rows. Per 32-key step: 8 WMMA for Q*K^T, online
// softmax (16-lane shuffle reductions), P->LDS->A-frag, 8 WMMA for P*V.
// =====================================================================
__global__ __launch_bounds__(256) void attn_kernel(
    const bf16_t* __restrict__ qb, const bf16_t* __restrict__ kb,
    const bf16_t* __restrict__ vb, bf16_t* __restrict__ yb)
{
  __shared__ __align__(16) bf16_t Ks[32][128];     // [key][dim]
  __shared__ __align__(16) bf16_t Vts[128][40];    // [dim][key] (transposed)
  __shared__ __align__(16) bf16_t Ps[8][16][32];   // per-wave P patch

  const int tid  = threadIdx.x;
  const int wid  = tid >> 5;
  const int lane = tid & 31;
  const int nlo  = lane & 15;
  const int half = lane >> 4;

  const int qt = blockIdx.x & 3;
  const int h  = (blockIdx.x >> 2) & (NHEADS - 1);
  const int b  = blockIdx.x >> 6;
  const size_t bh = (size_t)(b * NHEADS + h) * SEQ * HEADD;

  const int qrow0 = qt * 128 + wid * 16;           // wave's first query row

  // Q fragments stay resident: 4 chunks of 32 dims.
  v16bf qf[4];
  for (int c = 0; c < 4; ++c)
    qf[c] = load_frag16(qb + bh + (size_t)qrow0 * HEADD + c * 32, HEADD);

  v8f acc[8];
  for (int j = 0; j < 8; ++j) acc[j] = (v8f)(0.f);
  float rm[8], rl[8];
  for (int r = 0; r < 8; ++r) { rm[r] = -3.0e38f; rl[r] = 0.f; }

  const int kend = qt * 128 + 128;                 // causal block limit
  for (int k0 = 0; k0 < kend; k0 += 32) {
    // ---- stage K (row-major) and V (transposed), 2 x v8bf per thread ----
    for (int i = 0; i < 2; ++i) {
      int idx = tid + i * 256;                     // 0..511
      int kr  = idx >> 4;
      int c8  = idx & 15;
      *(v8bf*)&Ks[kr][c8 * 8] =
          *(const v8bf*)(kb + bh + (size_t)(k0 + kr) * HEADD + c8 * 8);
    }
    for (int i = 0; i < 2; ++i) {
      int idx = tid + i * 256;
      int kr  = idx >> 4;
      int c8  = idx & 15;
      v8bf vv = *(const v8bf*)(vb + bh + (size_t)(k0 + kr) * HEADD + c8 * 8);
      for (int e = 0; e < 8; ++e) Vts[c8 * 8 + e][kr] = vv[e];
    }
    __syncthreads();

    if (k0 <= qrow0 + 15) {                        // wave has live keys
      // ---- scores: 2 key tiles x 4 dim chunks ----
      v8f sc[2];
      sc[0] = (v8f)(0.f); sc[1] = (v8f)(0.f);
      for (int nt = 0; nt < 2; ++nt)
        for (int c = 0; c < 4; ++c) {
          v16bf bfr = load_frag16(&Ks[nt * 16][c * 32], 128);
          sc[nt] = wmma_bf16(qf[c], bfr, sc[nt]);
        }

      // ---- causal mask + running max ----
      float mnew[8];
      for (int r = 0; r < 8; ++r) {
        int qg = qrow0 + r + half * 8;
        float mx = rm[r];
        for (int nt = 0; nt < 2; ++nt) {
          int kg = k0 + nt * 16 + nlo;
          float v = sc[nt][r];
          if (kg > qg) v = -3.0e38f;
          sc[nt][r] = v;
          mx = fmaxf(mx, v);
        }
        for (int m = 1; m < 16; m <<= 1)
          mx = fmaxf(mx, __shfl_xor(mx, m, 32));
        mnew[r] = mx;
      }

      // ---- exp, row-sum, rescale accumulators ----
      for (int r = 0; r < 8; ++r) {
        float f = __expf(rm[r] - mnew[r]);
        float psum = 0.f;
        for (int nt = 0; nt < 2; ++nt) {
          float p = __expf(sc[nt][r] - mnew[r]);
          sc[nt][r] = p;
          psum += p;
        }
        for (int m = 1; m < 16; m <<= 1)
          psum += __shfl_xor(psum, m, 32);
        rl[r] = rl[r] * f + psum;
        rm[r] = mnew[r];
        for (int j = 0; j < 8; ++j) acc[j][r] *= f;
      }

      // ---- C-frag -> A-frag relayout of P via per-wave LDS patch ----
      for (int r = 0; r < 8; ++r) {
        int row = r + half * 8;
        Ps[wid][row][nlo]      = (bf16_t)sc[0][r];
        Ps[wid][row][16 + nlo] = (bf16_t)sc[1][r];
      }
      v16bf pf = load_frag16(&Ps[wid][0][0], 32);  // same-wave LDS: in-order

      // ---- O += P @ V ----
      for (int j = 0; j < 8; ++j) {
        v16bf bfr = load_frag16(&Vts[j * 16][0], 40);
        acc[j] = wmma_bf16(pf, bfr, acc[j]);
      }
    }
    __syncthreads();
  }

  // ---- normalize and write y as bf16 [B,T, H*hd] ----
  for (int j = 0; j < 8; ++j) {
    int d = j * 16 + nlo;
    for (int r = 0; r < 8; ++r) {
      int t = qrow0 + r + half * 8;
      float v = acc[j][r] / rl[r];
      yb[((size_t)(b * SEQ + t)) * DIMC + h * HEADD + d] = (bf16_t)v;
    }
  }
}

// =====================================================================
// Kernel 3: output projection  Y[8192,2048](bf16) @ Wout[2048,2048](fp32)
// -> out fp32. Same 128x128x32 WMMA tiling.
// =====================================================================
__global__ __launch_bounds__(256) void outproj_kernel(
    const bf16_t* __restrict__ Y, const float* __restrict__ W,
    float* __restrict__ O)
{
  const int K = DIMC, N = DIMC;
  __shared__ __align__(16) bf16_t As[128][40];
  __shared__ __align__(16) bf16_t Bs[128][40];

  const int tid  = threadIdx.x;
  const int wid  = tid >> 5;
  const int lane = tid & 31;
  const int wy = wid & 3, wx = wid >> 2;
  const int row0 = blockIdx.y * 128;
  const int col0 = blockIdx.x * 128;

  v8f acc[2][4];
  for (int i = 0; i < 2; ++i)
    for (int j = 0; j < 4; ++j)
      acc[i][j] = (v8f)(0.f);

  for (int k0 = 0; k0 < K; k0 += 32) {
    // ---- stage A: already bf16, 2 x v8bf per thread ----
    for (int i = 0; i < 2; ++i) {
      int idx = tid + i * 256;            // 0..511
      int r   = idx >> 2;                 // 4 x 8-elt runs per 32-wide row
      int c8  = idx & 3;
      *(v8bf*)&As[r][c8 * 8] =
          *(const v8bf*)(Y + (size_t)(row0 + r) * K + k0 + c8 * 8);
    }
    // ---- stage B: 32x128 fp32 -> bf16 transposed ----
    for (int i = 0; i < 4; ++i) {
      int idx = tid + i * 256;
      int kr  = idx >> 5;
      int c4  = idx & 31;
      float4 f = *(const float4*)(W + (size_t)(k0 + kr) * N + col0 + c4 * 4);
      Bs[c4 * 4 + 0][kr] = (bf16_t)f.x;
      Bs[c4 * 4 + 1][kr] = (bf16_t)f.y;
      Bs[c4 * 4 + 2][kr] = (bf16_t)f.z;
      Bs[c4 * 4 + 3][kr] = (bf16_t)f.w;
    }
    __syncthreads();

    v16bf a0 = load_frag16(&As[wy * 32][0],      40);
    v16bf a1 = load_frag16(&As[wy * 32 + 16][0], 40);
    for (int j = 0; j < 4; ++j) {
      v16bf b = load_frag16(&Bs[wx * 64 + j * 16][0], 40);
      acc[0][j] = wmma_bf16(a0, b, acc[0][j]);
      acc[1][j] = wmma_bf16(a1, b, acc[1][j]);
    }
    __syncthreads();
  }

  const int nlo  = lane & 15;
  const int half = lane >> 4;
  for (int i = 0; i < 2; ++i)
    for (int j = 0; j < 4; ++j) {
      int n = col0 + wx * 64 + j * 16 + nlo;
      for (int r = 0; r < 8; ++r) {
        int mg = row0 + wy * 32 + i * 16 + r + half * 8;
        O[(size_t)mg * N + n] = acc[i][j][r];
      }
    }
}

// =====================================================================
extern "C" void kernel_launch(void* const* d_in, const int* in_sizes, int n_in,
                              void* d_out, int out_size, void* d_ws, size_t ws_size,
                              hipStream_t stream) {
  (void)in_sizes; (void)n_in; (void)out_size; (void)ws_size;
  const float* x    = (const float*)d_in[0];   // [16,512,2048]
  const float* Wqkv = (const float*)d_in[1];   // [2048,6144]
  const float* Wout = (const float*)d_in[2];   // [2048,2048]
  float* out = (float*)d_out;                  // [16,512,2048]

  const size_t elems = (size_t)NBATCH * NHEADS * SEQ * HEADD;  // 16.7M
  bf16_t* qb = (bf16_t*)d_ws;
  bf16_t* kb = qb + elems;
  bf16_t* vb = kb + elems;
  bf16_t* yb = vb + elems;                     // [B*T, 2048] bf16

  dim3 blk(256);
  qkv_rope_kernel<<<dim3(3 * DIMC / 128, NBATCH * SEQ / 128), blk, 0, stream>>>(
      x, Wqkv, qb, kb, vb);
  attn_kernel<<<dim3(NBATCH * NHEADS * (SEQ / 128)), blk, 0, stream>>>(
      qb, kb, vb, yb);
  outproj_kernel<<<dim3(DIMC / 128, NBATCH * SEQ / 128), blk, 0, stream>>>(
      yb, Wout, out);
}
